// GNN_36893769072799
// MI455X (gfx1250) — matrix operand order repaired and checked by reading, not compiled, not run on hardware
//
#include <hip/hip_runtime.h>

#define N_NODES 100000
#define N_EDGES 1600000
#define IN_DIM 64
#define EMB 128
#define H1D 64
#define H2D 128
#define OUTD 10

typedef __bf16 bf16;
typedef __attribute__((ext_vector_type(8)))  bf16  v8bf;
typedef __attribute__((ext_vector_type(16))) bf16  v16bf;
typedef __attribute__((ext_vector_type(8)))  float v8f;

__device__ __forceinline__ v16bf frag16(const bf16* p0, const bf16* p1) {
  v8bf lo = *(const v8bf*)p0;
  v8bf hi = *(const v8bf*)p1;
  return __builtin_shufflevector(lo, hi, 0,1,2,3,4,5,6,7,8,9,10,11,12,13,14,15);
}

__device__ __forceinline__ v8f splat8(float b) {
  v8f r;
#pragma unroll
  for (int i = 0; i < 8; ++i) r[i] = b;
  return r;
}

// A source: per-wave LDS buffer, 16 rows x 128 cols (row stride 128 bf16).
// B source: LDS weights stored transposed: wt[n*K + k]  (n = output col, k contiguous).
template<int K, int NT>
__device__ __forceinline__ void gemm_tiles(const bf16* a0, const bf16* wt, int lane, v8f* acc) {
  const int m  = lane & 15;
  const int ah = (lane >> 4) << 3;   // +8  K-offset for upper half-wave (A layout)
  const int bh = (lane >> 4) << 3;   // +8  K-offset for upper half-wave (B layout)
  const bf16* arow = a0 + m * 128;
#pragma unroll
  for (int kk = 0; kk < K / 32; ++kk) {
    const int k0 = kk * 32;
    v16bf a = frag16(arow + k0 + ah, arow + k0 + ah + 16);
#pragma unroll
    for (int c = 0; c < NT; ++c) {
      const bf16* wr = wt + (c * 16 + m) * K + k0 + bh;
      v16bf b = frag16(wr, wr + 16);
      acc[c] = __builtin_amdgcn_wmma_f32_16x16x32_bf16(
          false, a, false, b, (short)0, acc[c], false, false);
    }
  }
}

__global__ void gnn_zero(float* __restrict__ p, int n) {
  int i = blockIdx.x * blockDim.x + threadIdx.x;
  if (i < n) p[i] = 0.0f;
}

// One edge per wave; 32 lanes cover the 64-dim feature row (2 floats each).
__global__ void gnn_scatter(const float* __restrict__ x, const int* __restrict__ ei,
                            float* __restrict__ agg, float* __restrict__ deg) {
  long long gtid = (long long)blockIdx.x * blockDim.x + threadIdx.x;
  int e    = (int)(gtid >> 5);
  int lane = (int)(gtid & 31);
  if (e >= N_EDGES) return;
  int s = ei[e];
  int d = ei[N_EDGES + e];
  float2 v = ((const float2*)(x + (size_t)s * IN_DIM))[lane];
  float* ap = agg + (size_t)d * IN_DIM + lane * 2;
  atomicAdd(ap,     v.x);
  atomicAdd(ap + 1, v.y);
  if (lane == 0) atomicAdd(deg + d, 1.0f);
}

__global__ void __launch_bounds__(128)
gnn_fused(const float* __restrict__ x,  const float* __restrict__ agg, const float* __restrict__ deg,
          const float* __restrict__ Wl, const float* __restrict__ bl,  const float* __restrict__ Wr,
          const float* __restrict__ W1, const float* __restrict__ b1,
          const float* __restrict__ W2, const float* __restrict__ b2,
          const float* __restrict__ W3, const float* __restrict__ b3,
          float* __restrict__ out) {
  __shared__ __align__(16) bf16 wbuf[128 * 128];     // current-stage weights, transposed
  __shared__ __align__(16) bf16 abuf[4][16][128];    // per-wave activation tile
  __shared__ float sB[336];

  const int tid  = threadIdx.x;
  const int w    = tid >> 5;
  const int lane = tid & 31;
  const int tile = blockIdx.x * 4 + w;
  const int tbase = tile * 16;

  // ---- biases ----
  for (int i = tid; i < 336; i += 128) {
    float v;
    if (i < 128)      v = bl[i];
    else if (i < 192) v = b1[i - 128];
    else if (i < 320) v = b2[i - 192];
    else if (i < 330) v = b3[i - 320];
    else              v = 0.0f;
    sB[i] = v;
  }
  // ---- stage-1 weights: Wc^T[n][k]; k<64 -> Wl[k][n], else Wr[k-64][n] ----
  for (int i = tid; i < 128 * 128; i += 128) {
    int n = i >> 7, k = i & 127;
    float v = (k < 64) ? Wl[k * EMB + n] : Wr[(k - 64) * EMB + n];
    wbuf[i] = (bf16)v;
  }
  __syncthreads();

  // ---- stage A input: [mean | x] as 16x128 bf16 ----
  {
    int r = lane >> 1, half = lane & 1;
    int node = tbase + r;
    if (node >= N_NODES) node = N_NODES - 1;
    bf16* arow = &abuf[w][r][0];
    if (half == 0) {
      float dg = deg[node];
      float rd = 1.0f / fmaxf(dg, 1.0f);
      const float4* p = (const float4*)(agg + (size_t)node * IN_DIM);
#pragma unroll
      for (int j = 0; j < 16; ++j) {
        float4 v = p[j];
        arow[4*j+0] = (bf16)(v.x * rd); arow[4*j+1] = (bf16)(v.y * rd);
        arow[4*j+2] = (bf16)(v.z * rd); arow[4*j+3] = (bf16)(v.w * rd);
      }
    } else {
      const float4* p = (const float4*)(x + (size_t)node * IN_DIM);
#pragma unroll
      for (int j = 0; j < 16; ++j) {
        float4 v = p[j];
        arow[64+4*j+0] = (bf16)v.x; arow[64+4*j+1] = (bf16)v.y;
        arow[64+4*j+2] = (bf16)v.z; arow[64+4*j+3] = (bf16)v.w;
      }
    }
  }
  __syncthreads();

  const int nn = lane & 15;
  const int mb = (lane >> 4) * 8;

  // ---- stage 1: h0 = [mean|x] @ Wc + b_l  (16x128, K=128) ----
  v8f acc[8];
#pragma unroll
  for (int c = 0; c < 8; ++c) acc[c] = splat8(sB[c * 16 + nn]);
  gemm_tiles<128, 8>(&abuf[w][0][0], wbuf, lane, acc);
  __syncthreads();
#pragma unroll
  for (int c = 0; c < 8; ++c)
#pragma unroll
    for (int r = 0; r < 8; ++r)
      abuf[w][mb + r][c * 16 + nn] = (bf16)acc[c][r];
  for (int i = tid; i < 64 * 128; i += 128) {       // W1^T[n][k], n<64, k<128
    int n = i >> 7, k = i & 127;
    wbuf[i] = (bf16)W1[k * H1D + n];
  }
  __syncthreads();

  // ---- stage 2: h1 = relu(h0 @ W1 + b1)  (16x64, K=128) ----
  v8f acc1[4];
#pragma unroll
  for (int c = 0; c < 4; ++c) acc1[c] = splat8(sB[128 + c * 16 + nn]);
  gemm_tiles<128, 4>(&abuf[w][0][0], wbuf, lane, acc1);
  __syncthreads();
#pragma unroll
  for (int c = 0; c < 4; ++c)
#pragma unroll
    for (int r = 0; r < 8; ++r)
      abuf[w][mb + r][c * 16 + nn] = (bf16)fmaxf(acc1[c][r], 0.0f);
  for (int i = tid; i < 128 * 64; i += 128) {       // W2^T[n][k], n<128, k<64
    int n = i >> 6, k = i & 63;
    wbuf[i] = (bf16)W2[k * H2D + n];
  }
  __syncthreads();

  // ---- stage 3: h2 = relu(h1 @ W2 + b2)  (16x128, K=64) ----
  v8f acc2[8];
#pragma unroll
  for (int c = 0; c < 8; ++c) acc2[c] = splat8(sB[192 + c * 16 + nn]);
  gemm_tiles<64, 8>(&abuf[w][0][0], wbuf, lane, acc2);
  __syncthreads();
#pragma unroll
  for (int c = 0; c < 8; ++c)
#pragma unroll
    for (int r = 0; r < 8; ++r)
      abuf[w][mb + r][c * 16 + nn] = (bf16)fmaxf(acc2[c][r], 0.0f);
  for (int i = tid; i < 16 * 128; i += 128) {       // W3^T[n][k] padded to 16 cols
    int n = i >> 7, k = i & 127;
    wbuf[i] = (bf16)((n < OUTD) ? W3[k * OUTD + n] : 0.0f);
  }
  __syncthreads();

  // ---- stage 4: out = h2 @ W3 + b3  (16x10, K=128) ----
  v8f acc3 = splat8((nn < OUTD) ? sB[320 + nn] : 0.0f);
  gemm_tiles<128, 1>(&abuf[w][0][0], wbuf, lane, &acc3);

  if (nn < OUTD) {
#pragma unroll
    for (int r = 0; r < 8; ++r) {
      int node = tbase + mb + r;
      if (node < N_NODES) out[(size_t)node * OUTD + nn] = acc3[r];
    }
  }
}

extern "C" void kernel_launch(void* const* d_in, const int* in_sizes, int n_in,
                              void* d_out, int out_size, void* d_ws, size_t ws_size,
                              hipStream_t stream) {
  (void)in_sizes; (void)n_in; (void)out_size; (void)ws_size;
  const float* x  = (const float*)d_in[0];
  const int*   ei = (const int*)  d_in[1];
  const float* Wl = (const float*)d_in[2];
  const float* bl = (const float*)d_in[3];
  const float* Wr = (const float*)d_in[4];
  const float* W1 = (const float*)d_in[5];
  const float* b1 = (const float*)d_in[6];
  const float* W2 = (const float*)d_in[7];
  const float* b2 = (const float*)d_in[8];
  const float* W3 = (const float*)d_in[9];
  const float* b3 = (const float*)d_in[10];
  float* out = (float*)d_out;

  float* agg = (float*)d_ws;                      // N_NODES * IN_DIM
  float* deg = agg + (size_t)N_NODES * IN_DIM;    // N_NODES

  const int ztot = N_NODES * IN_DIM + N_NODES;
  gnn_zero<<<(ztot + 255) / 256, 256, 0, stream>>>(agg, ztot);

  const long long sthreads = (long long)N_EDGES * 32;
  gnn_scatter<<<(int)((sthreads + 255) / 256), 256, 0, stream>>>(x, ei, agg, deg);

  const int tiles  = (N_NODES + 15) / 16;
  const int blocks = (tiles + 3) / 4;
  gnn_fused<<<blocks, 128, 0, stream>>>(x, agg, deg, Wl, bl, Wr,
                                        W1, b1, W2, b2, W3, b3, out);
}
